// mLSTMCell_29162827940155
// MI455X (gfx1250) — matrix operand order repaired and checked by reading, not compiled, not run on hardware
//
#include <hip/hip_runtime.h>
#include <hip/hip_bf16.h>

typedef __attribute__((ext_vector_type(16))) __bf16 v16bf;
typedef __attribute__((ext_vector_type(8)))  float  v8f;

typedef __attribute__((ext_vector_type(4))) unsigned int tdm_u32x4;
typedef __attribute__((ext_vector_type(8))) int          tdm_i32x8;
typedef __attribute__((ext_vector_type(4))) int          tdm_i32x4;

#define S_LEN 2048
#define E_DIM 1024
#define NHEAD 8
#define DHEAD 128
#define BR    128          // rows per workgroup tile
#define BC    128          // cols per iteration tile
#define EPS   1e-6f
#define LNEPS 1e-5f
#define INV_SQRT_DH 0.08838834764831845f   // 1/sqrt(128)

// ---------------------------------------------------------------------------
// TDM 2D tile load: 128x128 bf16 tile, dim0 contiguous in global, row stride
// `stride_elems` (elements).  D# packing per CDNA5 ISA ch.8:
//   group0: [1:0]=count=1 | lds_addr | global_addr[56:0] | type=2
//   group1: data_size=1(2B) | tensor_dim0/1=128 | tile_dim0/1=128 | dim0_stride
//   group2: tensor_dim2=1 (unused dims); group3: zeros
// ---------------------------------------------------------------------------
__device__ __forceinline__ void tdm_load_tile(unsigned lds_addr,
                                              const __bf16* gptr,
                                              int stride_elems)
{
    unsigned long long ga = (unsigned long long)(uintptr_t)gptr;
    tdm_u32x4 g0 = { 1u,
                     lds_addr,
                     (unsigned)ga,
                     (unsigned)((ga >> 32) & 0x01FFFFFFull) | (2u << 30) };
    tdm_i32x8 g1 = { 1 << 16,          // data_size = 1 (2 bytes)
                     BC << 16,         // tensor_dim0[15:0]
                     BC << 16,         // tensor_dim1[15:0]
                     BC << 16,         // tile_dim0
                     BC,               // tile_dim1 (tile_dim2 = 0)
                     stride_elems,     // tensor_dim0_stride[31:0]
                     0, 0 };
    tdm_i32x4 g2 = { 1, 0, 0, 0 };     // tensor_dim2 = 1 (unused)
    tdm_i32x4 g3 = { 0, 0, 0, 0 };
#if __clang_major__ >= 23
    tdm_i32x8 g4 = { 0, 0, 0, 0, 0, 0, 0, 0 };
    __builtin_amdgcn_tensor_load_to_lds(g0, g1, g2, g3, g4, 0);
#else
    __builtin_amdgcn_tensor_load_to_lds(g0, g1, g2, g3, 0);
#endif
}

// ---------------------------------------------------------------------------
// Kernel 0: precision/layout prep.  qb,kb = bf16 row-major copies of q,k.
// vtb = bf16 per-head transposed V:  vtb[(h*DH+d)*S + s] = v[s, h*DH+d].
// Output-coalesced mapping; bandwidth-bound (~36MB total traffic).
// ---------------------------------------------------------------------------
__global__ __launch_bounds__(256) void convert_kernel(
    const float* __restrict__ q, const float* __restrict__ k,
    const float* __restrict__ v,
    __bf16* __restrict__ qb, __bf16* __restrict__ kb, __bf16* __restrict__ vtb)
{
    int idx = blockIdx.x * 256 + threadIdx.x;        // 0 .. S*E-1
    qb[idx] = (__bf16)q[idx];
    kb[idx] = (__bf16)k[idx];
    int c = idx / S_LEN;                             // channel  (h*DH+d)
    int s = idx - c * S_LEN;                         // sequence position
    vtb[idx] = (__bf16)v[(size_t)s * E_DIM + c];
}

// ---------------------------------------------------------------------------
// Kernel 1: gate pre-activations.  grid = S, block = 256 (8 waves = 8 heads).
// ---------------------------------------------------------------------------
__global__ __launch_bounds__(256) void gates_kernel(
    const float* __restrict__ q, const float* __restrict__ k,
    const float* __restrict__ v,
    const float* __restrict__ igw, const float* __restrict__ igb,
    const float* __restrict__ fgw, const float* __restrict__ fgb,
    float* __restrict__ ig_pre, float* __restrict__ fg_pre)
{
    const int s    = blockIdx.x;
    const int h    = threadIdx.x >> 5;     // wave index == head (NH == 8)
    const int lane = threadIdx.x & 31;

    float si = 0.f, sf = 0.f;
    const float* qr = q + (size_t)s * E_DIM;
    const float* kr = k + (size_t)s * E_DIM;
    const float* vr = v + (size_t)s * E_DIM;
    const float* wi = igw + (size_t)h * 3 * E_DIM;
    const float* wf = fgw + (size_t)h * 3 * E_DIM;

    for (int idx = lane; idx < 3 * E_DIM; idx += 32) {
        float x;
        if (idx < E_DIM)            x = qr[idx];
        else if (idx < 2 * E_DIM)   x = kr[idx - E_DIM];
        else                        x = vr[idx - 2 * E_DIM];
        si += x * wi[idx];
        sf += x * wf[idx];
    }
    #pragma unroll
    for (int m = 16; m >= 1; m >>= 1) {
        si += __shfl_xor(si, m, 32);
        sf += __shfl_xor(sf, m, 32);
    }
    if (lane == 0) {
        ig_pre[h * S_LEN + s] = si + igb[h];
        fg_pre[h * S_LEN + s] = sf + fgb[h];
    }
}

// ---------------------------------------------------------------------------
// Kernel 2: cs[h,s] = inclusive cumsum_s logsigmoid(fg_pre[h,s]).
// ---------------------------------------------------------------------------
__global__ __launch_bounds__(32) void cumsum_kernel(
    const float* __restrict__ fg_pre, float* __restrict__ cs)
{
    const int h    = blockIdx.x;
    const int lane = threadIdx.x;
    float carry = 0.f;
    for (int base = 0; base < S_LEN; base += 32) {
        float x  = fg_pre[h * S_LEN + base + lane];
        float lf = fminf(x, 0.f) - log1pf(__expf(-fabsf(x)));
        #pragma unroll
        for (int off = 1; off < 32; off <<= 1) {
            float t = __shfl_up(lf, off, 32);
            if (lane >= off) lf += t;
        }
        cs[h * S_LEN + base + lane] = lf + carry;
        carry += __shfl(lf, 31, 32);
    }
}

// ---------------------------------------------------------------------------
// Kernel 3: flash-style mLSTM core.  grid = NH * (S/BR), block = 256 (8 waves).
// TDM double-buffered K/V tiles; WMMA bf16 GEMMs with f32 accumulation.
// ---------------------------------------------------------------------------
__global__ __launch_bounds__(256, 1) void mlstm_flash(
    const __bf16* __restrict__ qb, const __bf16* __restrict__ kb,
    const __bf16* __restrict__ vtb,
    const float* __restrict__ ig, const float* __restrict__ cs,
    const float* __restrict__ out_w, float* __restrict__ out)
{
    extern __shared__ char smem[];
    __bf16* sKb[2]; __bf16* sVb[2];
    sKb[0] = (__bf16*)smem;                    // [BC][DHEAD]
    sKb[1] = sKb[0] + BC * DHEAD;
    sVb[0] = sKb[1] + BC * DHEAD;              // [DHEAD][BC] (pre-transposed)
    sVb[1] = sVb[0] + BC * DHEAD;
    __bf16* sQP = sVb[1] + BC * DHEAD;         // [BR][DHEAD] Q tile, then P staging
    float*  sCs = (float*)(sQP + (size_t)BR * DHEAD);
    float*  sIg = sCs + BC;

    const int head = blockIdx.x / (S_LEN / BR);
    const int ib   = blockIdx.x % (S_LEN / BR);
    const int i0   = ib * BR;
    const int tid  = threadIdx.x;
    const int wave = tid >> 5;
    const int lane = tid & 31;
    const int lo   = lane & 15;
    const int hi   = lane >> 4;

    // ---- TDM preload: Q tile + first K/V tiles -------------------------------
    if (tid == 0) {
        tdm_load_tile((unsigned)(uintptr_t)sQP,
                      qb + (size_t)i0 * E_DIM + head * DHEAD, E_DIM);
        tdm_load_tile((unsigned)(uintptr_t)sKb[0],
                      kb + (size_t)0 * E_DIM + head * DHEAD, E_DIM);
        tdm_load_tile((unsigned)(uintptr_t)sVb[0],
                      vtb + (size_t)(head * DHEAD) * S_LEN + 0, S_LEN);
    }

    // ---- running flash state -------------------------------------------------
    v8f zero = {};
    v8f acc[8];
    v16bf qA[4];
    float mrow[8], brow[8], csi[8];
    #pragma unroll
    for (int d = 0; d < 8; ++d) acc[d] = zero;
    #pragma unroll
    for (int r = 0; r < 8; ++r) {
        mrow[r] = -INFINITY;
        brow[r] = 0.f;
        csi[r]  = cs[head * S_LEN + i0 + wave * 16 + r + 8 * hi];
    }

    __bf16* sPw = sQP + wave * 16 * BC;   // per-wave P staging (own Q rows only)

    for (int jb = 0; jb <= ib; ++jb) {
        const int j0 = jb * BC;
        const bool diag = (jb == ib);
        const __bf16* sK = sKb[jb & 1];
        const __bf16* sV = sVb[jb & 1];

        // prefetch next tile pair into the other buffer, then wait for current
        if (tid == 0) {
            if (jb < ib) {
                tdm_load_tile((unsigned)(uintptr_t)sKb[(jb + 1) & 1],
                              kb + (size_t)(j0 + BC) * E_DIM + head * DHEAD, E_DIM);
                tdm_load_tile((unsigned)(uintptr_t)sVb[(jb + 1) & 1],
                              vtb + (size_t)(head * DHEAD) * S_LEN + (j0 + BC), S_LEN);
                __builtin_amdgcn_s_wait_tensorcnt(2);   // current tiles complete
            } else {
                __builtin_amdgcn_s_wait_tensorcnt(0);
            }
        }
        if (tid < BC) {
            sCs[tid] = cs[head * S_LEN + j0 + tid];
            sIg[tid] = ig[head * S_LEN + j0 + tid];
        }
        __syncthreads();

        // ---- build Q A-fragments once (K = (e>>3)*16 + hi*8 + (e&7)) --------
        if (jb == 0) {
            const __bf16* rowp = sQP + (wave * 16 + lo) * DHEAD;
            #pragma unroll
            for (int c = 0; c < 4; ++c) {
                v16bf a;
                ((uint4*)&a)[0] = *(const uint4*)(rowp + c * 32 + hi * 8);
                ((uint4*)&a)[1] = *(const uint4*)(rowp + c * 32 + 16 + hi * 8);
                qA[c] = a;
            }
        }

        // ---- GEMM1: st[nt] = Q(16x128) * K^T tile(128x16) -------------------
        v8f st[8];
        #pragma unroll
        for (int nt = 0; nt < 8; ++nt) {
            v8f c = zero;
            #pragma unroll
            for (int c4 = 0; c4 < 4; ++c4) {
                const __bf16* kbp = sK + (nt * 16 + lo) * DHEAD + c4 * 32 + hi * 8;
                v16bf b;
                ((uint4*)&b)[0] = *(const uint4*)(kbp);
                ((uint4*)&b)[1] = *(const uint4*)(kbp + 16);
                c = __builtin_amdgcn_wmma_f32_16x16x32_bf16(
                        false, qA[c4], false, b, (short)0, c, false, false);
            }
            st[nt] = c;
        }

        // ---- gate matrix log_D + online max ---------------------------------
        float cj[8], gj[8];
        #pragma unroll
        for (int nt = 0; nt < 8; ++nt) {
            cj[nt] = sCs[nt * 16 + lo];
            gj[nt] = sIg[nt * 16 + lo];
        }
        float bm[8];
        #pragma unroll
        for (int r = 0; r < 8; ++r) bm[r] = -INFINITY;
        #pragma unroll
        for (int nt = 0; nt < 8; ++nt) {
            #pragma unroll
            for (int r = 0; r < 8; ++r) {
                float lv = csi[r] - cj[nt] + gj[nt];
                if (diag && (nt * 16 + lo > wave * 16 + r + 8 * hi)) lv = -INFINITY;
                bm[r] = fmaxf(bm[r], lv);
            }
        }
        #pragma unroll
        for (int m = 1; m < 16; m <<= 1) {
            #pragma unroll
            for (int r = 0; r < 8; ++r)
                bm[r] = fmaxf(bm[r], __shfl_xor(bm[r], m, 32));
        }

        float mnew[8], scl[8];
        #pragma unroll
        for (int r = 0; r < 8; ++r) {
            mnew[r] = fmaxf(mrow[r], bm[r]);
            scl[r]  = __expf(mrow[r] - mnew[r]);   // 0 when mrow was -inf
            mrow[r] = mnew[r];
            brow[r] *= scl[r];
        }
        #pragma unroll
        for (int d = 0; d < 8; ++d) {
            #pragma unroll
            for (int r = 0; r < 8; ++r) acc[d][r] *= scl[r];
        }

        // ---- P = exp(log_D - m) * qk/sqrt(dh); stage bf16; row sums ---------
        float bsum[8];
        #pragma unroll
        for (int r = 0; r < 8; ++r) bsum[r] = 0.f;
        #pragma unroll
        for (int nt = 0; nt < 8; ++nt) {
            #pragma unroll
            for (int r = 0; r < 8; ++r) {
                float lv = csi[r] - cj[nt] + gj[nt];
                float a  = st[nt][r] * INV_SQRT_DH;
                float p  = __expf(lv - mnew[r]) * a;
                if (diag && (nt * 16 + lo > wave * 16 + r + 8 * hi)) p = 0.f;
                bsum[r] += p;
                sPw[(r + 8 * hi) * BC + nt * 16 + lo] = (__bf16)p;
            }
        }
        #pragma unroll
        for (int m = 1; m < 16; m <<= 1) {
            #pragma unroll
            for (int r = 0; r < 8; ++r) bsum[r] += __shfl_xor(bsum[r], m, 32);
        }
        #pragma unroll
        for (int r = 0; r < 8; ++r) brow[r] += bsum[r];

        // ---- GEMM2: acc += P(16x128) * V(128x16) ----------------------------
        v16bf pA[4];
        {
            const __bf16* rowp = sPw + lo * BC;
            #pragma unroll
            for (int c = 0; c < 4; ++c) {
                v16bf a;
                ((uint4*)&a)[0] = *(const uint4*)(rowp + c * 32 + hi * 8);
                ((uint4*)&a)[1] = *(const uint4*)(rowp + c * 32 + 16 + hi * 8);
                pA[c] = a;
            }
        }
        #pragma unroll
        for (int dt = 0; dt < 8; ++dt) {
            v8f c = acc[dt];
            #pragma unroll
            for (int c4 = 0; c4 < 4; ++c4) {
                const __bf16* vbp = sV + (dt * 16 + lo) * BC + c4 * 32 + hi * 8;
                v16bf b;
                ((uint4*)&b)[0] = *(const uint4*)(vbp);
                ((uint4*)&b)[1] = *(const uint4*)(vbp + 16);
                c = __builtin_amdgcn_wmma_f32_16x16x32_bf16(
                        false, pA[c4], false, b, (short)0, c, false, false);
            }
            acc[dt] = c;
        }
        __syncthreads();   // all reads done before next TDM overwrites buffers
    }

    // ---- normalizer + fused group-norm + store -------------------------------
    float nrm[8], sum[8], ssq[8];
    #pragma unroll
    for (int r = 0; r < 8; ++r) {
        nrm[r] = fmaxf(fabsf(brow[r]), __expf(-mrow[r])) + EPS;
        sum[r] = 0.f;
        ssq[r] = 0.f;
    }
    #pragma unroll
    for (int dt = 0; dt < 8; ++dt) {
        #pragma unroll
        for (int r = 0; r < 8; ++r) {
            float hv = acc[dt][r] / nrm[r];
            acc[dt][r] = hv;
            sum[r] += hv;
            ssq[r] += hv * hv;
        }
    }
    #pragma unroll
    for (int m = 1; m < 16; m <<= 1) {
        #pragma unroll
        for (int r = 0; r < 8; ++r) {
            sum[r] += __shfl_xor(sum[r], m, 32);
            ssq[r] += __shfl_xor(ssq[r], m, 32);
        }
    }
    #pragma unroll
    for (int dt = 0; dt < 8; ++dt) {
        float w = out_w[head * DHEAD + dt * 16 + lo];
        #pragma unroll
        for (int r = 0; r < 8; ++r) {
            float mean = sum[r] * (1.0f / DHEAD);
            float var  = ssq[r] * (1.0f / DHEAD) - mean * mean;
            float rstd = rsqrtf(var + LNEPS);
            float val  = (acc[dt][r] - mean) * rstd * w;
            int row = i0 + wave * 16 + r + 8 * hi;
            out[(size_t)row * E_DIM + head * DHEAD + dt * 16 + lo] = val;
        }
    }
}

// ---------------------------------------------------------------------------
extern "C" void kernel_launch(void* const* d_in, const int* in_sizes, int n_in,
                              void* d_out, int out_size, void* d_ws, size_t ws_size,
                              hipStream_t stream) {
    const float* q   = (const float*)d_in[0];
    const float* k   = (const float*)d_in[1];
    const float* v   = (const float*)d_in[2];
    const float* igw = (const float*)d_in[3];
    const float* igb = (const float*)d_in[4];
    const float* fgw = (const float*)d_in[5];
    const float* fgb = (const float*)d_in[6];
    const float* ow  = (const float*)d_in[7];

    char* wsb = (char*)d_ws;
    float* ig_pre = (float*)wsb;                               // NH*S f32
    float* fg_pre = ig_pre + NHEAD * S_LEN;                    // NH*S f32
    float* cs_buf = fg_pre + NHEAD * S_LEN;                    // NH*S f32
    __bf16* qb  = (__bf16*)(wsb + 3 * NHEAD * S_LEN * sizeof(float));
    __bf16* kbf = qb  + (size_t)S_LEN * E_DIM;
    __bf16* vtb = kbf + (size_t)S_LEN * E_DIM;

    convert_kernel<<<(S_LEN * E_DIM) / 256, 256, 0, stream>>>(q, k, v, qb, kbf, vtb);
    gates_kernel<<<S_LEN, 256, 0, stream>>>(q, k, v, igw, igb, fgw, fgb,
                                            ig_pre, fg_pre);
    cumsum_kernel<<<NHEAD, 32, 0, stream>>>(fg_pre, cs_buf);

    size_t shmem = 5 * (size_t)BC * DHEAD * sizeof(__bf16) + 2 * BC * sizeof(float);
    (void)hipFuncSetAttribute((const void*)mlstm_flash,
                              hipFuncAttributeMaxDynamicSharedMemorySize,
                              (int)shmem);
    mlstm_flash<<<NHEAD * (S_LEN / BR), 256, shmem, stream>>>(
        qb, kbf, vtb, ig_pre, cs_buf, ow, (float*)d_out);
}
